// LSTM_22565758174072
// MI455X (gfx1250) — matrix operand order repaired and checked by reading, not compiled, not run on hardware
//
#include <hip/hip_runtime.h>
#include <hip/hip_bf16.h>

// ---------------------------------------------------------------------------
// LSTM for MI455X (gfx1250): B=32, T=1024, D=1024, H=128 (gates 4H=512)
// Phase 1: gx = x @ W_ih^T + (b_ih + b_hh)  -- big WMMA GEMM (bf16x3 split)
// Phase 2: persistent single-workgroup recurrence, W_hh fragments in VGPRs,
//          double-buffered gate slices prefetched via async global->LDS.
// ---------------------------------------------------------------------------

typedef __bf16 bf16_t;
typedef __attribute__((ext_vector_type(16))) __bf16 v16bf;
typedef __attribute__((ext_vector_type(8)))  __bf16 v8bf;
typedef __attribute__((ext_vector_type(8)))  float  v8f;
typedef __attribute__((ext_vector_type(4)))  int    v4i;

#define WMMA_BF16(A, B, C) \
  __builtin_amdgcn_wmma_f32_16x16x32_bf16(false, (A), false, (B), (short)0, (C), false, false)

#if __has_builtin(__builtin_amdgcn_global_load_async_to_lds_b128)
#define HAVE_ASYNC_LDS 1
#else
#define HAVE_ASYNC_LDS 0
#endif

#if __has_builtin(__builtin_amdgcn_s_wait_asynccnt)
#define WAIT_ASYNC(n) __builtin_amdgcn_s_wait_asynccnt(n)
#else
#define WAIT_ASYNC(n) asm volatile("s_wait_asynccnt %0" ::"i"(n) : "memory")
#endif

static __device__ __forceinline__ v16bf cat8(v8bf a, v8bf b) {
  v16bf r;
#pragma unroll
  for (int i = 0; i < 8; ++i) { r[i] = a[i]; r[i + 8] = b[i]; }
  return r;
}

// A fragment (16x32 bf16): source row-major, row stride `ld` elements.
// Lane layout: m = lane%16; 16 contiguous K starting at 16g, g = lane/16.
static __device__ __forceinline__ v16bf load_a_frag(const bf16_t* base, int ld, int lane) {
  const int m = lane & 15, g = lane >> 4;
  const bf16_t* p = base + m * ld + 16 * g;
  v8bf a0 = *(const v8bf*)p;        // 16B aligned
  v8bf a1 = *(const v8bf*)(p + 8);  // 16B aligned
  return cat8(a0, a1);
}

// B fragment (32x16 bf16) from weights stored [n][k] row-major (= B column-major).
// Lane layout: n = lane%16; 16 contiguous K starting at 16*(lane/16).
static __device__ __forceinline__ v16bf load_b_frag(const bf16_t* base, int ld, int lane) {
  const int n = lane & 15, g = lane >> 4;
  const bf16_t* p = base + n * ld + 16 * g;
  v8bf b0 = *(const v8bf*)p;
  v8bf b1 = *(const v8bf*)(p + 8);
  return cat8(b0, b1);
}

static __device__ __forceinline__ float sigmoid_f(float x) {
  return 1.f / (1.f + __expf(-x));
}
// Stable fast tanh: saturates to +/-1 without inf/inf NaN.
static __device__ __forceinline__ float tanh_f(float x) {
  return 1.f - 2.f / (1.f + __expf(2.f * x));
}

// ---------------------------------------------------------------------------
// Phase 1: gx[m, n] = sum_d x[m, d] * W_ih[n, d] + b_ih[n] + b_hh[n]
// m = b*T + t in [0, 32768), n in [0, 512). Block tile 64x64, 8 waves,
// each wave: one 16-row M-subtile x two 16-col N-subtiles. bf16 hi/lo split.
// ---------------------------------------------------------------------------
__global__ __launch_bounds__(256)
void lstm_gate_gemm(const float* __restrict__ x, const float* __restrict__ W_ih,
                    const float* __restrict__ b_ih, const float* __restrict__ b_hh,
                    float* __restrict__ gx)
{
  __shared__ __align__(16) bf16_t As_hi[64][32];
  __shared__ __align__(16) bf16_t As_lo[64][32];
  __shared__ __align__(16) bf16_t Bs_hi[64][32];
  __shared__ __align__(16) bf16_t Bs_lo[64][32];

  const int tid  = threadIdx.x;
  const int lane = tid & 31;
  const int wave = tid >> 5;
  const int mi   = wave & 3;          // M-subtile 0..3
  const int ni0  = (wave >> 2) * 2;   // first of two N-subtiles
  const int m_base = blockIdx.x * 64;
  const int n_base = blockIdx.y * 64;

  v8f acc0 = {};
  v8f acc1 = {};

  for (int kb = 0; kb < 32; ++kb) {
    const int k0 = kb * 32;
    // Stage 64x32 tiles of x and W_ih as hi/lo bf16 (8 contiguous f32 per thread each)
#pragma unroll
    for (int i = 0; i < 8; ++i) {
      const int l = tid * 8 + i;           // 0..2047
      const int r = l >> 5, c = l & 31;
      const float fa = x[(size_t)(m_base + r) * 1024 + k0 + c];
      const bf16_t ah = (bf16_t)fa;
      As_hi[r][c] = ah;
      As_lo[r][c] = (bf16_t)(fa - (float)ah);
      const float fb = W_ih[(size_t)(n_base + r) * 1024 + k0 + c];
      const bf16_t bh = (bf16_t)fb;
      Bs_hi[r][c] = bh;
      Bs_lo[r][c] = (bf16_t)(fb - (float)bh);
    }
    __syncthreads();

    const v16bf Ah = load_a_frag(&As_hi[mi * 16][0], 32, lane);
    const v16bf Al = load_a_frag(&As_lo[mi * 16][0], 32, lane);
    {
      const v16bf Bh = load_b_frag(&Bs_hi[ni0 * 16][0], 32, lane);
      const v16bf Bl = load_b_frag(&Bs_lo[ni0 * 16][0], 32, lane);
      acc0 = WMMA_BF16(Ah, Bh, acc0);
      acc0 = WMMA_BF16(Al, Bh, acc0);
      acc0 = WMMA_BF16(Ah, Bl, acc0);
    }
    {
      const v16bf Bh = load_b_frag(&Bs_hi[(ni0 + 1) * 16][0], 32, lane);
      const v16bf Bl = load_b_frag(&Bs_lo[(ni0 + 1) * 16][0], 32, lane);
      acc1 = WMMA_BF16(Ah, Bh, acc1);
      acc1 = WMMA_BF16(Al, Bh, acc1);
      acc1 = WMMA_BF16(Ah, Bl, acc1);
    }
    __syncthreads();
  }

  // Epilogue: C/D layout -> row = v + 8*(lane/16), col = lane%16
  const int col = lane & 15, rg = lane >> 4;
  {
    const int n = n_base + ni0 * 16 + col;
    const float bias = b_ih[n] + b_hh[n];
#pragma unroll
    for (int v = 0; v < 8; ++v) {
      const int m = m_base + mi * 16 + rg * 8 + v;
      gx[(size_t)m * 512 + n] = acc0[v] + bias;
    }
  }
  {
    const int n = n_base + (ni0 + 1) * 16 + col;
    const float bias = b_ih[n] + b_hh[n];
#pragma unroll
    for (int v = 0; v < 8; ++v) {
      const int m = m_base + mi * 16 + rg * 8 + v;
      gx[(size_t)m * 512 + n] = acc1[v] + bias;
    }
  }
}

// Stage gate slice gx[:, t, :] (32x512 f32) into an LDS buffer.
// 4 x b128 transfers per thread; async path uses GLOBAL_LOAD_ASYNC_TO_LDS_B128.
static __device__ __forceinline__ void stage_gates(const float* __restrict__ gx, int t,
                                                   float* __restrict__ dst, int tid)
{
#pragma unroll
  for (int i = 0; i < 4; ++i) {
    const int idx4 = i * 1024 + tid;      // 0..4095 float4s
    const int b = idx4 >> 7, n4 = idx4 & 127;
    const float* src = &gx[((size_t)b * 1024 + t) * 512 + (size_t)n4 * 4];
    float* d = &dst[b * 512 + n4 * 4];
#if HAVE_ASYNC_LDS
    __builtin_amdgcn_global_load_async_to_lds_b128(
        (__attribute__((address_space(1))) v4i*)src,
        (__attribute__((address_space(3))) v4i*)d, 0, 0);
#else
    *(float4*)d = *(const float4*)src;
#endif
  }
}

// ---------------------------------------------------------------------------
// Phase 2: persistent recurrence. One workgroup, 1024 threads = 32 waves.
// Wave w owns gate columns [16w, 16w+16) x both batch tiles (b 0-15, 16-31).
// W_hh bf16 hi/lo B-fragments live in registers for the whole kernel.
// Gate slices double-buffered; step t+1 prefetched async during step t.
// ---------------------------------------------------------------------------
__global__ __launch_bounds__(1024, 1)
void lstm_recurrence(const float* __restrict__ gx, const float* __restrict__ W_hh,
                     float* __restrict__ out, float* __restrict__ h_n,
                     float* __restrict__ c_n)
{
  extern __shared__ char smem[];
  float*  g_buf0 = (float*)smem;                     // [32][512] = 64 KB
  float*  g_buf1 = g_buf0 + 32 * 512;                // [32][512] = 64 KB
  bf16_t* h_hi   = (bf16_t*)(g_buf1 + 32 * 512);     // [32][128] =  8 KB
  bf16_t* h_lo   = h_hi + 32 * 128;                  // [32][128] =  8 KB

  const int tid  = threadIdx.x;
  const int lane = tid & 31;
  const int wave = tid >> 5;
  const int n0   = wave * 16;

  // Preload W_hh (512x128 f32) as hi/lo bf16 B fragments: 4 k-blocks of 32.
  v16bf Bh[4], Bl[4];
  {
    const int n = n0 + (lane & 15);
    const int g = lane >> 4;
#pragma unroll
    for (int kb = 0; kb < 4; ++kb) {
      const float* p = W_hh + (size_t)n * 128 + kb * 32 + 16 * g;
      v16bf bh, bl;
#pragma unroll
      for (int i = 0; i < 16; ++i) {
        const float f  = p[i];
        const bf16_t hi = (bf16_t)f;
        bh[i] = hi;
        bl[i] = (bf16_t)(f - (float)hi);
      }
      Bh[kb] = bh; Bl[kb] = bl;
    }
  }

  float c_reg[4] = {0.f, 0.f, 0.f, 0.f};  // cell state, 4 (b,j) cells per thread

  for (int i = tid; i < 32 * 128; i += 1024) { h_hi[i] = (bf16_t)0.f; h_lo[i] = (bf16_t)0.f; }

  // Prefetch gate slice for t = 0.
  stage_gates(gx, 0, g_buf0, tid);

  __syncthreads();

  for (int t = 0; t < 1024; ++t) {
    float* gcur  = (t & 1) ? g_buf1 : g_buf0;
    float* gnext = (t & 1) ? g_buf0 : g_buf1;

    // Prefetch t+1 before compute so the async transfer overlaps the WMMAs,
    // then wait only for the transfers belonging to step t.
    if (t + 1 < 1024) {
      stage_gates(gx, t + 1, gnext, tid);
      WAIT_ASYNC(4);
    } else {
      WAIT_ASYNC(0);
    }
    __syncthreads();

    // gates += h_prev @ W_hh^T (bf16x3, K=128 in 4 blocks of 32)
#pragma unroll
    for (int mt = 0; mt < 2; ++mt) {
      v8f acc = {};
#pragma unroll
      for (int kb = 0; kb < 4; ++kb) {
        const v16bf Ah = load_a_frag(h_hi + (mt * 16) * 128 + kb * 32, 128, lane);
        const v16bf Al = load_a_frag(h_lo + (mt * 16) * 128 + kb * 32, 128, lane);
        acc = WMMA_BF16(Ah, Bh[kb], acc);
        acc = WMMA_BF16(Al, Bh[kb], acc);
        acc = WMMA_BF16(Ah, Bl[kb], acc);
      }
      const int col = lane & 15, rg = lane >> 4;
#pragma unroll
      for (int v = 0; v < 8; ++v) {
        const int b = mt * 16 + rg * 8 + v;
        gcur[b * 512 + n0 + col] += acc[v];  // each (b,n) owned by exactly one lane
      }
    }
    __syncthreads();

    // Gate nonlinearities + state update (4 cells per thread)
#pragma unroll
    for (int q = 0; q < 4; ++q) {
      const int l = q * 1024 + tid;          // 0..4095
      const int b = l >> 7, j = l & 127;
      const float gi = gcur[b * 512 +       j];
      const float gf = gcur[b * 512 + 128 + j];
      const float gg = gcur[b * 512 + 256 + j];
      const float go = gcur[b * 512 + 384 + j];
      const float i_ = sigmoid_f(gi);
      const float f_ = sigmoid_f(gf);
      const float g_ = tanh_f(gg);
      const float o_ = sigmoid_f(go);
      const float c  = f_ * c_reg[q] + i_ * g_;
      c_reg[q] = c;
      const float h = o_ * tanh_f(c);
      const bf16_t hh = (bf16_t)h;
      h_hi[b * 128 + j] = hh;
      h_lo[b * 128 + j] = (bf16_t)(h - (float)hh);
      out[(size_t)b * (1024 * 128) + (size_t)t * 128 + j] = h;
      if (t == 1023) { h_n[b * 128 + j] = h; c_n[b * 128 + j] = c; }
    }
    __syncthreads();
  }
}

extern "C" void kernel_launch(void* const* d_in, const int* in_sizes, int n_in,
                              void* d_out, int out_size, void* d_ws, size_t ws_size,
                              hipStream_t stream) {
  (void)in_sizes; (void)n_in; (void)out_size; (void)ws_size;
  const float* x    = (const float*)d_in[0];
  const float* W_ih = (const float*)d_in[1];
  const float* W_hh = (const float*)d_in[2];
  const float* b_ih = (const float*)d_in[3];
  const float* b_hh = (const float*)d_in[4];

  float* out = (float*)d_out;
  float* h_n = out + (size_t)32 * 1024 * 128;
  float* c_n = h_n + (size_t)32 * 128;
  float* gx  = (float*)d_ws;  // 32768 x 512 f32 = 64 MB

  // Phase 1: 512 M-tiles x 8 N-tiles, 256 threads each
  lstm_gate_gemm<<<dim3(512, 8), dim3(256), 0, stream>>>(x, W_ih, b_ih, b_hh, gx);

  // Phase 2: one persistent workgroup, 144 KB dynamic LDS
  const size_t shmem = 2 * 32 * 512 * 4 + 2 * 32 * 128 * 2;  // 147456 B
  (void)hipFuncSetAttribute((const void*)lstm_recurrence,
                            hipFuncAttributeMaxDynamicSharedMemorySize, (int)shmem);
  lstm_recurrence<<<dim3(1), dim3(1024), shmem, stream>>>(gx, W_hh, out, h_n, c_n);
}
